// SelfAttention_88648124989898
// MI455X (gfx1250) — matrix operand order repaired and checked by reading, not compiled, not run on hardware
//
#include <hip/hip_runtime.h>

typedef __attribute__((ext_vector_type(16))) _Float16 v16h;
typedef __attribute__((ext_vector_type(8)))  _Float16 v8h;
typedef __attribute__((ext_vector_type(8)))  float    v8f;

#define WMMA16(A,B,C) __builtin_amdgcn_wmma_f32_16x16x32_f16(false,(A),false,(B),(short)0,(C),false,false)

#define B_  4
#define C_  512
#define N_  4096
#define D_  64

__device__ __forceinline__ v16h mk16(v8h a, v8h b) {
    v16h r;
#pragma unroll
    for (int i = 0; i < 8; ++i) { r[i] = a[i]; r[i + 8] = b[i]; }
    return r;
}

// ---------------- elementwise f32 -> f16 convert ----------------
__global__ void cvt_f32_f16(const float* __restrict__ in, _Float16* __restrict__ out, int n) {
    int i = blockIdx.x * blockDim.x + threadIdx.x;
    if (i < n) out[i] = (_Float16)in[i];
}

// ---------------- x[B,C,N] f32 -> xt[B,N,C] f16 (tiled transpose) ----------------
__global__ __launch_bounds__(256) void transpose_cvt(const float* __restrict__ x,
                                                     _Float16* __restrict__ xt) {
    __shared__ _Float16 tile[32][33];
    const int b  = blockIdx.z;
    const int ct = blockIdx.y;   // C/32
    const int nt = blockIdx.x;   // N/32
    const int tx = threadIdx.x;  // 0..31
    const int ty = threadIdx.y;  // 0..7
    const float* xb = x + ((size_t)b * C_ + (size_t)ct * 32) * N_ + (size_t)nt * 32;
#pragma unroll
    for (int k = 0; k < 4; ++k)
        tile[ty + 8 * k][tx] = (_Float16)xb[(size_t)(ty + 8 * k) * N_ + tx];
    __syncthreads();
    _Float16* xo = xt + ((size_t)b * N_ + (size_t)nt * 32) * C_ + (size_t)ct * 32;
#pragma unroll
    for (int k = 0; k < 4; ++k)
        xo[(size_t)(ty + 8 * k) * C_ + tx] = tile[tx][ty + 8 * k];
}

// ---------------- WMMA NT-GEMM: Out[m,n] = sum_c A[m,c]*Bm[n,c] + bias ----------------
// A: [M,512] f16 row-major; Bm: [Nn,512] f16 row-major; Out: [M,ldo] f16.
// One wave computes a 16x64 tile; software-pipelined fragment loads (1 K-step ahead).
__global__ __launch_bounds__(256) void proj_gemm(const _Float16* __restrict__ A,
                                                 const _Float16* __restrict__ Bm,
                                                 _Float16* __restrict__ Out,
                                                 const float* __restrict__ bias,
                                                 int M, int Nt, int ldo, int bias_by_row) {
    const int lane = threadIdx.x & 31;
    const int w    = threadIdx.x >> 5;
    const int lm   = lane & 15;
    const int hi   = lane >> 4;
    const int ntile64 = Nt >> 6;
    const int wt = blockIdx.x * 8 + w;
    if (wt >= (M >> 4) * ntile64) return;    // wave-uniform guard (EXEC stays all-ones)
    const int tn = wt % ntile64;             // 64-wide column tile
    const int tm = wt / ntile64;

    const _Float16* arow = A + (size_t)(tm * 16 + lm) * 512;
    const _Float16* brow[4];
#pragma unroll
    for (int j = 0; j < 4; ++j)
        brow[j] = Bm + (size_t)(tn * 64 + j * 16 + lm) * 512;

    v8f acc[4];
#pragma unroll
    for (int j = 0; j < 4; ++j) acc[j] = (v8f){};

    // prologue: fragments for kb = 0
    v16h af, bf[4];
    {
        v8h a0 = *(const v8h*)(arow + 16 * hi);
        v8h a1 = *(const v8h*)(arow + 8 + 16 * hi);
        af = mk16(a0, a1);
#pragma unroll
        for (int j = 0; j < 4; ++j)
            bf[j] = *(const v16h*)(brow[j] + 16 * hi);
    }
#pragma unroll 3
    for (int kn = 32; kn < 512; kn += 32) {
        // load next K-step into fresh registers (overlaps with WMMAs below)
        v16h afn, bfn[4];
        {
            v8h a0 = *(const v8h*)(arow + kn + 16 * hi);
            v8h a1 = *(const v8h*)(arow + kn + 8 + 16 * hi);
            afn = mk16(a0, a1);
#pragma unroll
            for (int j = 0; j < 4; ++j)
                bfn[j] = *(const v16h*)(brow[j] + kn + 16 * hi);
        }
#pragma unroll
        for (int j = 0; j < 4; ++j) acc[j] = WMMA16(af, bf[j], acc[j]);
        af = afn;
#pragma unroll
        for (int j = 0; j < 4; ++j) bf[j] = bfn[j];
    }
    // epilogue K-step
#pragma unroll
    for (int j = 0; j < 4; ++j) acc[j] = WMMA16(af, bf[j], acc[j]);

#pragma unroll
    for (int j = 0; j < 4; ++j) {
        const int ncol = tn * 64 + j * 16 + lm;
#pragma unroll
        for (int r = 0; r < 8; ++r) {
            const int mrow = tm * 16 + r + 8 * hi;
            float v = acc[j][r] + bias[bias_by_row ? mrow : ncol];
            Out[(size_t)mrow * ldo + ncol] = (_Float16)v;
        }
    }
}

// ---------------- flash attention + epilogue ----------------
// grid: B * N/32 blocks; block: 256 threads (8 waves).
// Workgroup handles 32 query rows (2 i-tiles); wave w owns channels [64w, 64w+64):
// 2x4 accumulator tiles. Each V B-fragment feeds both i-tiles (2 WMMA per load).
__global__ __launch_bounds__(256) void attn_kernel(const _Float16* __restrict__ Qt,  // [B,N,64]
                                                   const _Float16* __restrict__ Kt,  // [B,N,64]
                                                   const _Float16* __restrict__ Vh,  // [B,C,N]
                                                   const float*    __restrict__ x,   // [B,C,N]
                                                   const float*    __restrict__ gamma_p,
                                                   float*          __restrict__ out) {
    __shared__ _Float16 Plds[32][136];   // row stride 272B (16B aligned, bank-skewed)
    __shared__ float pm[8][32];
    __shared__ float ps[8][32];
    __shared__ float m_state[32];
    __shared__ float l_state[32];

    const int tid  = threadIdx.x;
    const int w    = tid >> 5;
    const int lane = tid & 31;
    const int lm   = lane & 15;
    const int hi   = lane >> 4;

    const int b     = blockIdx.x >> 7;       // N/32 = 128 tiles per batch
    const int ibase = (blockIdx.x & 127) << 5;

    const _Float16* Qb = Qt + (size_t)b * N_ * D_;
    const _Float16* Kb = Kt + (size_t)b * N_ * D_;
    const _Float16* Vb = Vh + (size_t)b * C_ * N_;

    if (tid < 32) { m_state[tid] = -1e30f; l_state[tid] = 0.0f; }
    __syncthreads();

    // Q A-fragments: 2 i-tiles x 2 K-chunks (d=0..31, d=32..63), loaded once
    v16h qa[2][2];
#pragma unroll
    for (int ii = 0; ii < 2; ++ii) {
        const _Float16* qrow = Qb + (size_t)(ibase + ii * 16 + lm) * D_;
        v8h a0 = *(const v8h*)(qrow +  0 + 16 * hi);
        v8h a1 = *(const v8h*)(qrow +  8 + 16 * hi);
        v8h a2 = *(const v8h*)(qrow + 32 + 16 * hi);
        v8h a3 = *(const v8h*)(qrow + 40 + 16 * hi);
        qa[ii][0] = mk16(a0, a1);
        qa[ii][1] = mk16(a2, a3);
    }

    v8f acc[2][4];
#pragma unroll
    for (int ii = 0; ii < 2; ++ii)
#pragma unroll
        for (int t = 0; t < 4; ++t) acc[ii][t] = (v8f){};
    const int cw = w * 64;

    for (int jb = 0; jb < N_; jb += 128) {
        // prefetch next j-step's K rows and V stream into cache
        if (jb + 128 < N_) {
            __builtin_prefetch(Kb + (size_t)(jb + 128 + w * 16 + lm) * D_, 0, 1);
            __builtin_prefetch(Vb + (size_t)(cw + lane) * N_ + jb + 128, 0, 1);
            __builtin_prefetch(Vb + (size_t)(cw + 32 + lane) * N_ + jb + 128, 0, 1);
        }

        // ---- S tiles: this wave's 16-key subtile x 2 i-tiles ----
        const _Float16* krow = Kb + (size_t)(jb + w * 16 + lm) * D_;
        v16h kb0 = *(const v16h*)(krow +  0 + 16 * hi);   // d = 0..31
        v16h kb1 = *(const v16h*)(krow + 32 + 16 * hi);   // d = 32..63
        v8f s[2];
#pragma unroll
        for (int ii = 0; ii < 2; ++ii) {
            v8f t = {};
            t = WMMA16(qa[ii][0], kb0, t);
            t = WMMA16(qa[ii][1], kb1, t);
            s[ii] = t;
        }

        // ---- local row max (16-lane butterfly within each half-wave) ----
#pragma unroll
        for (int ii = 0; ii < 2; ++ii) {
#pragma unroll
            for (int r = 0; r < 8; ++r) {
                float v = s[ii][r];
                v = fmaxf(v, __shfl_xor(v, 1, 32));
                v = fmaxf(v, __shfl_xor(v, 2, 32));
                v = fmaxf(v, __shfl_xor(v, 4, 32));
                v = fmaxf(v, __shfl_xor(v, 8, 32));
                if (lm == 0) pm[w][ii * 16 + r + 8 * hi] = v;
            }
        }
        __syncthreads();                                   // B1

        // ---- new running max + rescale factor (redundant per lane, deterministic) ----
        float mnew[2][8], scale[2][8];
#pragma unroll
        for (int ii = 0; ii < 2; ++ii)
#pragma unroll
            for (int r = 0; r < 8; ++r) {
                const int row = ii * 16 + r + 8 * hi;
                const float mo = m_state[row];
                float mn = mo;
#pragma unroll
                for (int ww = 0; ww < 8; ++ww) mn = fmaxf(mn, pm[ww][row]);
                mnew[ii][r]  = mn;
                scale[ii][r] = __expf(mo - mn);
            }

        // ---- P = exp(S - mnew); partial row sums; stash P (f16) to LDS ----
#pragma unroll
        for (int ii = 0; ii < 2; ++ii)
#pragma unroll
            for (int r = 0; r < 8; ++r) {
                float p = __expf(s[ii][r] - mnew[ii][r]);
                Plds[ii * 16 + r + 8 * hi][w * 16 + lm] = (_Float16)p;
                float t = p;
                t += __shfl_xor(t, 1, 32);
                t += __shfl_xor(t, 2, 32);
                t += __shfl_xor(t, 4, 32);
                t += __shfl_xor(t, 8, 32);
                if (lm == 0) ps[w][ii * 16 + r + 8 * hi] = t;
            }

        // ---- rescale running accumulators ----
#pragma unroll
        for (int ii = 0; ii < 2; ++ii)
#pragma unroll
            for (int t = 0; t < 4; ++t)
#pragma unroll
                for (int r = 0; r < 8; ++r) acc[ii][t][r] *= scale[ii][r];

        __syncthreads();                                   // B2

        // ---- one lane per row commits the running stats (32 rows = 32 lanes) ----
        if (w == 0) {
            const int row = lane;
            const float mo = m_state[row];
            float mn = mo;
#pragma unroll
            for (int ww = 0; ww < 8; ++ww) mn = fmaxf(mn, pm[ww][row]);
            float l = l_state[row] * __expf(mo - mn);
#pragma unroll
            for (int ww = 0; ww < 8; ++ww) l += ps[ww][row];
            m_state[row] = mn;
            l_state[row] = l;
        }
        __syncthreads();                                   // B3

        // ---- O += P · V : 4 K-chunks; per chunk load all operands, then 8 WMMA ----
#pragma unroll
        for (int kk = 0; kk < 4; ++kk) {
            v16h pa[2];
#pragma unroll
            for (int ii = 0; ii < 2; ++ii) {
                const _Float16* pr = &Plds[ii * 16 + lm][kk * 32 + 16 * hi];
                pa[ii] = mk16(*(const v8h*)(pr), *(const v8h*)(pr + 8));
            }
            const _Float16* vbase = Vb + (size_t)jb + kk * 32 + 16 * hi;
            v16h vb[4];
#pragma unroll
            for (int t = 0; t < 4; ++t)
                vb[t] = *(const v16h*)(vbase + (size_t)(cw + t * 16 + lm) * N_);
#pragma unroll
            for (int t = 0; t < 4; ++t) {
                acc[0][t] = WMMA16(pa[0], vb[t], acc[0][t]);
                acc[1][t] = WMMA16(pa[1], vb[t], acc[1][t]);
            }
        }
    }

    // ---- epilogue: out = gamma * O/l + x ----
    const float g = *gamma_p;
    float rl[2][8];
#pragma unroll
    for (int ii = 0; ii < 2; ++ii)
#pragma unroll
        for (int r = 0; r < 8; ++r) rl[ii][r] = 1.0f / l_state[ii * 16 + r + 8 * hi];
    const float* xb = x   + (size_t)b * C_ * N_;
    float*       ob = out + (size_t)b * C_ * N_;
#pragma unroll
    for (int ii = 0; ii < 2; ++ii)
#pragma unroll
        for (int t = 0; t < 4; ++t) {
            const int c = cw + t * 16 + lm;
#pragma unroll
            for (int r = 0; r < 8; ++r) {
                const int i = ibase + ii * 16 + r + 8 * hi;
                const size_t idx = (size_t)c * N_ + i;
                ob[idx] = g * (acc[ii][t][r] * rl[ii][r]) + xb[idx];
            }
        }
}

extern "C" void kernel_launch(void* const* d_in, const int* in_sizes, int n_in,
                              void* d_out, int out_size, void* d_ws, size_t ws_size,
                              hipStream_t stream) {
    const float* x     = (const float*)d_in[0];
    const float* Wq    = (const float*)d_in[1];
    const float* bq    = (const float*)d_in[2];
    const float* Wk    = (const float*)d_in[3];
    const float* bk    = (const float*)d_in[4];
    const float* Wv    = (const float*)d_in[5];
    const float* bv    = (const float*)d_in[6];
    const float* gamma = (const float*)d_in[7];
    float* out = (float*)d_out;

    // workspace carve-up (all f16, sizes are multiples of 256B) — ~36.7 MB total
    char* ws = (char*)d_ws;
    size_t o = 0;
    _Float16* xt  = (_Float16*)(ws + o); o += (size_t)B_ * N_ * C_ * 2;   // [B,N,C]
    _Float16* Qt  = (_Float16*)(ws + o); o += (size_t)B_ * N_ * D_ * 2;   // [B,N,64]
    _Float16* Kt  = (_Float16*)(ws + o); o += (size_t)B_ * N_ * D_ * 2;   // [B,N,64]
    _Float16* Vh  = (_Float16*)(ws + o); o += (size_t)B_ * C_ * N_ * 2;   // [B,C,N]
    _Float16* Wqh = (_Float16*)(ws + o); o += (size_t)D_ * C_ * 2;
    _Float16* Wkh = (_Float16*)(ws + o); o += (size_t)D_ * C_ * 2;
    _Float16* Wvh = (_Float16*)(ws + o); o += (size_t)C_ * C_ * 2;
    (void)o; (void)ws_size; (void)in_sizes; (void)n_in; (void)out_size;

    // 1) weight converts
    cvt_f32_f16<<<(D_ * C_ + 255) / 256, 256, 0, stream>>>(Wq, Wqh, D_ * C_);
    cvt_f32_f16<<<(D_ * C_ + 255) / 256, 256, 0, stream>>>(Wk, Wkh, D_ * C_);
    cvt_f32_f16<<<(C_ * C_ + 255) / 256, 256, 0, stream>>>(Wv, Wvh, C_ * C_);

    // 2) x -> xt (transpose + f16)
    transpose_cvt<<<dim3(N_ / 32, C_ / 32, B_), dim3(32, 8), 0, stream>>>(x, xt);

    // 3) projections (WMMA GEMMs, 16x64 tile per wave, pipelined), per batch
    for (int b = 0; b < B_; ++b) {
        const _Float16* xtb = xt + (size_t)b * N_ * C_;
        // Qt[n,d] = xt[n,:]·Wq[d,:] + bq[d] : (4096/16)*(64/64) = 256 waves -> 32 blocks
        proj_gemm<<<32, 256, 0, stream>>>(xtb, Wqh, Qt + (size_t)b * N_ * D_, bq,
                                          N_, D_, D_, 0);
        proj_gemm<<<32, 256, 0, stream>>>(xtb, Wkh, Kt + (size_t)b * N_ * D_, bk,
                                          N_, D_, D_, 0);
        // V[c,n] = Wv[c,:]·xt[n,:] + bv[c] : (512/16)*(4096/64) = 2048 waves -> 256 blocks
        proj_gemm<<<256, 256, 0, stream>>>(Wvh, xtb, Vh + (size_t)b * C_ * N_, bv,
                                           C_, N_, N_, 1);
    }

    // 4) flash attention + epilogue (32 query rows per workgroup)
    attn_kernel<<<B_ * (N_ / 32), 256, 0, stream>>>(Qt, Kt, Vh, x, gamma, out);
}
